// EdgeEmb_34256659153228
// MI455X (gfx1250) — matrix operand order repaired and checked by reading, not compiled, not run on hardware
//
#include <hip/hip_runtime.h>

typedef __attribute__((ext_vector_type(16))) __bf16 v16bf;
typedef __attribute__((ext_vector_type(8)))  __bf16 bf16x8;
typedef __attribute__((ext_vector_type(4)))  __bf16 bf16x4;
typedef __attribute__((ext_vector_type(8)))  float  v8f;

#define WT_S 144                              // W1T/W2T row stride (128 K + 16 pad), 288B = 16B aligned
#define WM_S 272                              // WmT  row stride (256 K + 16 pad), 544B = 16B aligned
#define W1T_OFF 0
#define W2T_OFF (128 * WT_S * 2)              // 36864
#define WMT_OFF (W2T_OFF + 128 * WT_S * 2)    // 73728
#define STG_OFF (WMT_OFF + 256 * WM_S * 2)    // 212992
#define STG_PER_WAVE (16 * 256 * 2)           // 8192 B: [16 edges][256 ch] bf16
#define SMEM_BYTES (STG_OFF + 8 * STG_PER_WAVE) // 278528 < 320KB WGP LDS

static __device__ __forceinline__ __bf16 f2bf(float f) {
    unsigned u = __builtin_bit_cast(unsigned, f);
    u += 0x7FFFu + ((u >> 16) & 1u);                 // round-to-nearest-even
    unsigned short s = (unsigned short)(u >> 16);
    return __builtin_bit_cast(__bf16, s);
}

static __device__ __forceinline__ v16bf ld_frag2(const __bf16* p0, const __bf16* p1) {
    union { v16bf v; bf16x8 h[2]; } u;
    u.h[0] = *(const bf16x8*)p0;
    u.h[1] = *(const bf16x8*)p1;
    return u.v;
}
// A 16x32 bf16 fragment (ISA 7.12.2): lane half h holds K = kk+8h..+7 and kk+16+8h..+7
static __device__ __forceinline__ v16bf ld_a(const __bf16* row, int kk, int half) {
    const __bf16* p = row + kk + half * 8;
    return ld_frag2(p, p + 16);
}
// B 32x16 bf16 fragment from transposed weight (rows = N, K contiguous, stride s elems)
static __device__ __forceinline__ v16bf ld_b(const __bf16* wt, int s, int n, int kk, int half) {
    const __bf16* p = wt + n * s + kk + half * 8;
    return ld_frag2(p, p + 16);
}
static __device__ __forceinline__ v8f wmma_bf16(v16bf a, v16bf b, v8f c) {
    return __builtin_amdgcn_wmma_f32_16x16x32_bf16(false, a, false, b, (short)0, c, false, false);
}

__global__ __launch_bounds__(256, 1) void EdgeEmb_34256659153228_kernel(
    const float* __restrict__ node, const long long* __restrict__ ei,
    const float* __restrict__ W1, const float* __restrict__ b1,
    const float* __restrict__ W2, const float* __restrict__ b2,
    const float* __restrict__ Wm, const float* __restrict__ bm,
    const float* __restrict__ gamma, const float* __restrict__ beta,
    float* __restrict__ out, int n_edges) {
    extern __shared__ char smem[];
    __bf16* w1t = (__bf16*)(smem + W1T_OFF);
    __bf16* w2t = (__bf16*)(smem + W2T_OFF);
    __bf16* wmt = (__bf16*)(smem + WMT_OFF);

    const int tid  = threadIdx.x;
    const int lane = tid & 31;
    const int wid  = tid >> 5;
    const int l15  = lane & 15;
    const int half = lane >> 4;

    // Cooperative weight preload: f32 global -> bf16 transposed LDS (amortized over 128 edges)
    for (int i = tid; i < 128 * 128; i += 256) {
        int k = i >> 7, n = i & 127;
        w1t[n * WT_S + k] = f2bf(W1[i]);
        w2t[n * WT_S + k] = f2bf(W2[i]);
    }
    for (int i = tid; i < 256 * 256; i += 256) {
        int k = i >> 8, n = i & 255;
        wmt[n * WM_S + k] = f2bf(Wm[i]);
    }
    __syncthreads();

    __bf16* stage = (__bf16*)(smem + STG_OFF + wid * STG_PER_WAVE);
    const long long ebase = (long long)blockIdx.x * 128 + wid * 16;
    if (ebase >= n_edges) return;

    // ---- Gather: lanes 0-15 hold src index of edge l15, lanes 16-31 hold dst index
    long long eid = ebase + l15;
    if (eid >= n_edges) eid = n_edges - 1;
    long long nid = (half == 0) ? ei[eid] : ei[(long long)n_edges + eid];
#pragma unroll
    for (int row = 0; row < 32; ++row) {
        long long rn = __shfl(nid, row, 32);                       // v_readlane, imm lane
        float4 v = *(const float4*)(node + rn * 128 + lane * 4);   // coalesced 512B/row, L2 hit
        bf16x4 t = { f2bf(v.x), f2bf(v.y), f2bf(v.z), f2bf(v.w) };
        int m = row & 15;
        int cb = (row < 16) ? 0 : 128;                             // src cols 0-127, dst 128-255
        *(bf16x4*)(stage + m * 256 + cb + lane * 4) = t;
    }

    // ---- Phase 1: src_emb = src@W1, dst_emb = dst@W2  (64 WMMA)
    v8f zero = {};
    v8f accS[8], accD[8];
#pragma unroll
    for (int nt = 0; nt < 8; ++nt) { accS[nt] = zero; accD[nt] = zero; }

    const __bf16* arow = stage + l15 * 256;
#pragma unroll
    for (int ks = 0; ks < 4; ++ks) {
        const int kk = ks * 32;
        v16bf aS = ld_a(arow, kk, half);
        v16bf aD = ld_a(arow + 128, kk, half);
#pragma unroll
        for (int nt = 0; nt < 8; ++nt) {
            accS[nt] = wmma_bf16(aS, ld_b(w1t, WT_S, nt * 16 + l15, kk, half), accS[nt]);
            accD[nt] = wmma_bf16(aD, ld_b(w2t, WT_S, nt * 16 + l15, kk, half), accD[nt]);
        }
    }
    // bias + write edge_attr back to staging as bf16 (D layout: VGPR r, lane group -> edge m)
#pragma unroll
    for (int nt = 0; nt < 8; ++nt) {
        float bb1 = b1[nt * 16 + l15];
        float bb2 = b2[nt * 16 + l15];
#pragma unroll
        for (int r = 0; r < 8; ++r) {
            int m = r + half * 8;
            stage[m * 256 + nt * 16 + l15]       = f2bf(accS[nt][r] + bb1);
            stage[m * 256 + 128 + nt * 16 + l15] = f2bf(accD[nt][r] + bb2);
        }
    }

    // ---- Phase 2: h = edge_attr @ Wm (K=256), GeGLU fused per (x,gate) tile pair (128 WMMA)
    // Hoist all 8 A-fragments once (64 VGPRs); inner loop is pure B-load + WMMA.
    v16bf aF[8];
#pragma unroll
    for (int ks = 0; ks < 8; ++ks) aF[ks] = ld_a(arow, ks * 32, half);

    v8f yv[8];
    float s1[8], s2[8];
#pragma unroll
    for (int r = 0; r < 8; ++r) { s1[r] = 0.f; s2[r] = 0.f; }

#pragma unroll
    for (int ct = 0; ct < 8; ++ct) {
        v8f ax = zero, ag = zero;
#pragma unroll
        for (int ks = 0; ks < 8; ++ks) {
            const int kk = ks * 32;
            ax = wmma_bf16(aF[ks], ld_b(wmt, WM_S, ct * 16 + l15, kk, half), ax);
            ag = wmma_bf16(aF[ks], ld_b(wmt, WM_S, 128 + ct * 16 + l15, kk, half), ag);
        }
        float bx = bm[ct * 16 + l15];
        float bg = bm[128 + ct * 16 + l15];
#pragma unroll
        for (int r = 0; r < 8; ++r) {
            float xv = ax[r] + bx;
            float gv = ag[r] + bg;
            float y  = xv * (0.5f * gv * (1.0f + erff(gv * 0.70710678118654752f)));
            yv[ct][r] = y;
            s1[r] += y;
            s2[r] += y * y;
        }
    }

    // ---- LayerNorm: reduce over 16-lane groups (xor masks < 16 stay in-group)
    float mu[8], rs[8];
#pragma unroll
    for (int r = 0; r < 8; ++r) {
        float a = s1[r], b = s2[r];
#pragma unroll
        for (int off = 1; off < 16; off <<= 1) {
            a += __shfl_xor(a, off, 32);
            b += __shfl_xor(b, off, 32);
        }
        float m_ = a * (1.0f / 128.0f);
        float v_ = fmaxf(b * (1.0f / 128.0f) - m_ * m_, 0.0f);
        mu[r] = m_;
        rs[r] = rsqrtf(v_ + 1e-5f);
    }

#pragma unroll
    for (int ct = 0; ct < 8; ++ct) {
        float g  = gamma[ct * 16 + l15];
        float be = beta[ct * 16 + l15];
#pragma unroll
        for (int r = 0; r < 8; ++r) {
            int m = r + half * 8;
            long long e = ebase + m;
            if (e < n_edges)
                out[e * 128 + ct * 16 + l15] = (yv[ct][r] - mu[r]) * rs[r] * g + be;
        }
    }
}

extern "C" void kernel_launch(void* const* d_in, const int* in_sizes, int n_in,
                              void* d_out, int out_size, void* d_ws, size_t ws_size,
                              hipStream_t stream) {
    const float*     node  = (const float*)d_in[0];
    const long long* ei    = (const long long*)d_in[1];
    const float* W1 = (const float*)d_in[2];
    const float* b1 = (const float*)d_in[3];
    const float* W2 = (const float*)d_in[4];
    const float* b2 = (const float*)d_in[5];
    const float* Wm = (const float*)d_in[6];
    const float* bm = (const float*)d_in[7];
    const float* gm = (const float*)d_in[8];
    const float* bt = (const float*)d_in[9];
    float* out = (float*)d_out;

    const int n_edges = in_sizes[1] / 2;
    const int blocks  = (n_edges + 127) / 128;      // 16 edges/wave, 8 waves/block
    EdgeEmb_34256659153228_kernel<<<dim3(blocks), dim3(256), SMEM_BYTES, stream>>>(
        node, ei, W1, b1, W2, b2, Wm, bm, gm, bt, out, n_edges);
}
